// GeomBM_84911503442152
// MI455X (gfx1250) — matrix operand order repaired and checked by reading, not compiled, not run on hardware
//
#include <hip/hip_runtime.h>
#include <hip/hip_bf16.h>
#include <math.h>

#define NN   10000
#define EE   160000
#define HH   128
#define FF   256
#define RR   64
#define LL   3
#define CHUNK 16000

typedef __attribute__((ext_vector_type(16))) _Float16 v16h;
typedef __attribute__((ext_vector_type(8)))  _Float16 v8h;
typedef __attribute__((ext_vector_type(8)))  float    v8f;

__device__ __forceinline__ float siluf(float x) { return x / (1.f + __expf(-x)); }
__device__ __forceinline__ float sigmf(float x) { return 1.f / (1.f + __expf(-x)); }

// ---------------------------------------------------------------------------
// WMMA GEMM, 16x64 output per wave (A-fragment reuse across 4 N-subtiles):
//   C[M,Nc] = act(A[M,K](f16) * Bt[Nc,K](f16)^T + bias)
// K multiple of 32, M multiple of 16, Nc multiple of 64.
// A-fragment lane layout (16-bit A 16x32): lane L holds row M=L&15;
//   halves j=0..15 -> K = k0 + (L>=16?16:0) + j
// B-fragment: lane L holds col N=L&15; halves j=0..15 -> K = k0 + (L>=16?16:0) + j
// Both map to contiguous 16B loads per lane per operand.
// ---------------------------------------------------------------------------
__global__ void __launch_bounds__(256)
wmma_gemm_k(const _Float16* __restrict__ A, const _Float16* __restrict__ Bt,
            float* __restrict__ C, const float* __restrict__ bias,
            int M, int K, int Nc, int act)
{
    int lane = threadIdx.x & 31;
    int wave = threadIdx.x >> 5;
    int tilesN = Nc >> 6;                      // 64-wide N supertiles
    int totalTiles = (M >> 4) * tilesN;
    int tile = blockIdx.x * 8 + wave;
    if (tile >= totalTiles) return;
    int tm = tile / tilesN, tn = tile - tm * tilesN;
    int l16 = lane & 15;
    int hi  = lane >> 4;

    size_t arow = (size_t)(tm * 16 + l16) * (size_t)K;
    size_t brow0 = (size_t)(tn * 64 + l16) * (size_t)K;        // subtile 0
    // subtile s adds 16*K halves to the B row base
    v8f acc0 = {0.f,0.f,0.f,0.f,0.f,0.f,0.f,0.f};
    v8f acc1 = acc0, acc2 = acc0, acc3 = acc0;

    for (int k0 = 0; k0 < K; k0 += 32) {
        v8h a0 = *(const v8h*)(A + arow + k0 + hi * 16);
        v8h a1 = *(const v8h*)(A + arow + k0 + hi * 16 + 8);
        v16h av;
#pragma unroll
        for (int j = 0; j < 8; ++j) { av[j] = a0[j]; av[j + 8] = a1[j]; }

        size_t bk = brow0 + k0 + hi * 16;
        v8h b00 = *(const v8h*)(Bt + bk);
        v8h b01 = *(const v8h*)(Bt + bk + 8);
        v8h b10 = *(const v8h*)(Bt + bk + (size_t)16 * K);
        v8h b11 = *(const v8h*)(Bt + bk + (size_t)16 * K + 8);
        v8h b20 = *(const v8h*)(Bt + bk + (size_t)32 * K);
        v8h b21 = *(const v8h*)(Bt + bk + (size_t)32 * K + 8);
        v8h b30 = *(const v8h*)(Bt + bk + (size_t)48 * K);
        v8h b31 = *(const v8h*)(Bt + bk + (size_t)48 * K + 8);
        v16h bv0, bv1, bv2, bv3;
#pragma unroll
        for (int j = 0; j < 8; ++j) {
            bv0[j] = b00[j]; bv0[j + 8] = b01[j];
            bv1[j] = b10[j]; bv1[j + 8] = b11[j];
            bv2[j] = b20[j]; bv2[j + 8] = b21[j];
            bv3[j] = b30[j]; bv3[j + 8] = b31[j];
        }
        acc0 = __builtin_amdgcn_wmma_f32_16x16x32_f16(false, av, false, bv0,
                                                      (short)0, acc0, false, false);
        acc1 = __builtin_amdgcn_wmma_f32_16x16x32_f16(false, av, false, bv1,
                                                      (short)0, acc1, false, false);
        acc2 = __builtin_amdgcn_wmma_f32_16x16x32_f16(false, av, false, bv2,
                                                      (short)0, acc2, false, false);
        acc3 = __builtin_amdgcn_wmma_f32_16x16x32_f16(false, av, false, bv3,
                                                      (short)0, acc3, false, false);
    }

    int rowBase = tm * 16 + hi * 8;
#pragma unroll
    for (int s = 0; s < 4; ++s) {
        v8f acc = (s == 0) ? acc0 : (s == 1) ? acc1 : (s == 2) ? acc2 : acc3;
        int col = tn * 64 + s * 16 + l16;
        float bv2f = bias ? bias[col] : 0.f;
#pragma unroll
        for (int r = 0; r < 8; ++r) {
            float v = acc[r] + bv2f;
            if (act == 1) v = siluf(v);
            else if (act == 2) v = sigmf(v);
            C[(size_t)(rowBase + r) * (size_t)Nc + col] = v;
        }
    }
}

// ---------------------------------------------------------------------------
// Small utility kernels
// ---------------------------------------------------------------------------
__global__ void zero_f32_k(float* p, long long n) {
    long long i = (long long)blockIdx.x * blockDim.x + threadIdx.x;
    if (i < n) p[i] = 0.f;
}
__global__ void copy_f32_k(const float* __restrict__ a, float* __restrict__ b, long long n) {
    long long i = (long long)blockIdx.x * blockDim.x + threadIdx.x;
    if (i < n) b[i] = a[i];
}
__global__ void f32_to_f16_k(const float* __restrict__ a, _Float16* __restrict__ b, long long n) {
    long long i = (long long)blockIdx.x * blockDim.x + threadIdx.x;
    if (i < n) b[i] = (_Float16)a[i];
}
// W [K,Nc] f32 row-major  ->  Wt [Nc,K] f16 row-major
__global__ void trans_f16_k(const float* __restrict__ W, _Float16* __restrict__ Wt,
                            int K, int Nc) {
    int idx = blockIdx.x * blockDim.x + threadIdx.x;
    if (idx >= K * Nc) return;
    int k = idx / Nc, n = idx - k * Nc;
    Wt[(size_t)n * K + k] = (_Float16)W[idx];
}

// ---------------------------------------------------------------------------
// Edge feature precompute: rbf (f16), cosine cutoff, unit vectors
// ---------------------------------------------------------------------------
__global__ void edge_pre_k(const float* __restrict__ ew, const float* __restrict__ ev,
                           _Float16* __restrict__ rbfH, float* __restrict__ ccut,
                           float* __restrict__ uvec) {
    int e = blockIdx.x * blockDim.x + threadIdx.x;
    if (e >= EE) return;
    float d = ew[e];
    const float gamma = 0.1f * (64.f / 5.f) * (64.f / 5.f);   // 16.384
    _Float16* rr = rbfH + (size_t)e * RR;
    for (int i = 0; i < RR; ++i) {
        float mu = 5.f * (float)i / 63.f;
        float t = d - mu;
        rr[i] = (_Float16)__expf(-gamma * t * t);
    }
    float c = 0.5f * (__cosf(3.14159265358979f * d / 5.f) + 1.f);
    ccut[e] = (d < 5.f) ? c : 0.f;
    float x = ev[e * 3 + 0], y = ev[e * 3 + 1], z = ev[e * 3 + 2];
    float inv = 1.f / (sqrtf(x * x + y * y + z * z) + 1e-8f);
    uvec[e * 3 + 0] = x * inv; uvec[e * 3 + 1] = y * inv; uvec[e * 3 + 2] = z * inv;
}

// h = emb[z] + t * Wt
__global__ void node_init_k(const int* __restrict__ z, const float* __restrict__ t,
                            const float* __restrict__ emb, const float* __restrict__ wt,
                            float* __restrict__ h) {
    int idx = blockIdx.x * blockDim.x + threadIdx.x;
    if (idx >= NN * HH) return;
    int n = idx >> 7, c = idx & 127;
    h[idx] = emb[z[n] * HH + c] + t[n] * wt[c];
}

// LayerNorm -> f16 (one 128-thread block per node)
__global__ void ln_k(const float* __restrict__ h, _Float16* __restrict__ hnH) {
    __shared__ float s1[128], s2[128];
    int n = blockIdx.x, c = threadIdx.x;
    float x = h[(size_t)n * HH + c];
    s1[c] = x; s2[c] = x * x;
    __syncthreads();
    for (int st = 64; st > 0; st >>= 1) {
        if (c < st) { s1[c] += s1[c + st]; s2[c] += s2[c + st]; }
        __syncthreads();
    }
    float m = s1[0] * (1.f / 128.f);
    float v = s2[0] * (1.f / 128.f) - m * m;
    hnH[(size_t)n * HH + c] = (_Float16)((x - m) * rsqrtf(v + 1e-5f));
}

// ---------------------------------------------------------------------------
// Edge message kernel: one wave (32 lanes) per edge, 4 channels per lane.
// qkv row layout: [q 0:128 | k 128:256 | v 256:640]; dkv row: [dk 0:128 | dv 128:512]
// ---------------------------------------------------------------------------
__global__ void __launch_bounds__(256)
edge_msg_k(const int* __restrict__ ei, const float* __restrict__ ccut,
           const float* __restrict__ uvec, const float* __restrict__ qkv,
           const float* __restrict__ dkv, const float* __restrict__ vec,
           float* __restrict__ msgH, float* __restrict__ msgV,
           int c0e, int cnt) {
    int ew = blockIdx.x * 8 + (threadIdx.x >> 5);
    if (ew >= cnt) return;
    int lane = threadIdx.x & 31;
    int e = c0e + ew;
    int dst = ei[e], src = ei[EE + e];
    float cc = ccut[e];
    int ch0 = lane * 4;
    int hd = lane >> 2;                 // 8 heads, 16 channels each
    const float* qrow = qkv + (size_t)dst * 640;
    const float* krow = qkv + (size_t)src * 640 + 128;
    const float* vrow = qkv + (size_t)src * 640 + 256;
    const float* dkr = dkv + (size_t)ew * 512;
    const float* dvr = dkr + 128;
    float part = 0.f;
#pragma unroll
    for (int j = 0; j < 4; ++j) {
        int c = ch0 + j;
        part += qrow[c] * krow[c] * dkr[c];
    }
    part += __shfl_xor(part, 1, 4);
    part += __shfl_xor(part, 2, 4);
    float attn = siluf(part) * cc;
    const float* vs = vec + (size_t)src * 384;
    float ux = uvec[e * 3 + 0], uy = uvec[e * 3 + 1], uz = uvec[e * 3 + 2];
    float* mh = msgH + (size_t)dst * 128;
    float* mv = msgV + (size_t)dst * 384;
#pragma unroll
    for (int j = 0; j < 4; ++j) {
        int c = ch0 + j;
        int jj = c & 15;
        int vi = hd * 48 + jj;
        float xe  = vrow[vi]      * dvr[vi]      * attn;
        float v1f = vrow[vi + 16] * dvr[vi + 16];
        float v2f = vrow[vi + 32] * dvr[vi + 32];
        atomicAdd(&mh[c], xe);
        atomicAdd(&mv[c],       vs[c]       * v1f + ux * v2f);
        atomicAdd(&mv[128 + c], vs[128 + c] * v1f + uy * v2f);
        atomicAdd(&mv[256 + c], vs[256 + c] * v1f + uz * v2f);
    }
}

// node update: h += dot*o2 + o3 ; vec += msgV + v3p*o1
__global__ void node_upd_k(float* __restrict__ h, float* __restrict__ vec,
                           const float* __restrict__ vp, const float* __restrict__ o,
                           const float* __restrict__ msgV) {
    int idx = blockIdx.x * blockDim.x + threadIdx.x;
    if (idx >= NN * HH) return;
    int n = idx >> 7, c = idx & 127;
    const float* orow = o + (size_t)n * 384;
    float o1 = orow[c], o2 = orow[128 + c], o3 = orow[256 + c];
    float dot = 0.f;
#pragma unroll
    for (int i = 0; i < 3; ++i) {
        const float* vpr = vp + (size_t)(n * 3 + i) * 384;
        dot += vpr[c] * vpr[128 + c];
    }
    h[idx] += dot * o2 + o3;
#pragma unroll
    for (int i = 0; i < 3; ++i) {
        const float* vpr = vp + (size_t)(n * 3 + i) * 384;
        size_t vix = (size_t)n * 384 + i * 128 + c;
        vec[vix] += msgV[vix] + vpr[256 + c] * o1;
    }
}

// build f16 concat [h | sqrt(sum_i vh^2)]  rows of length 384
__global__ void cat_build_k(const float* __restrict__ h, const float* __restrict__ vh,
                            _Float16* __restrict__ catH) {
    int idx = blockIdx.x * blockDim.x + threadIdx.x;
    if (idx >= NN * 384) return;
    int n = idx / 384, c = idx - n * 384;
    if (c < 128) {
        catH[idx] = (_Float16)h[(size_t)n * 128 + c];
    } else {
        int f = c - 128;
        float a = vh[(size_t)(n * 3 + 0) * 256 + f];
        float b = vh[(size_t)(n * 3 + 1) * 256 + f];
        float d = vh[(size_t)(n * 3 + 2) * 256 + f];
        catH[idx] = (_Float16)sqrtf(a * a + b * b + d * d + 1e-8f);
    }
}

// vout_pre * gate, normalize by (1+|.|), reduce over channels -> pred[N,3]
__global__ void gvp_fin_k(const float* __restrict__ vout, const float* __restrict__ gate,
                          float* __restrict__ pred) {
    int idx = blockIdx.x * blockDim.x + threadIdx.x;
    if (idx >= NN * HH) return;
    int n = idx >> 7, c = idx & 127;
    float g = gate[(size_t)n * 128 + c];
    float v0 = vout[(size_t)(n * 3 + 0) * 128 + c] * g;
    float v1 = vout[(size_t)(n * 3 + 1) * 128 + c] * g;
    float v2 = vout[(size_t)(n * 3 + 2) * 128 + c] * g;
    float inv = 1.f / (1.f + sqrtf(v0 * v0 + v1 * v1 + v2 * v2 + 1e-8f));
    atomicAdd(&pred[n * 3 + 0], v0 * inv);
    atomicAdd(&pred[n * 3 + 1], v1 * inv);
    atomicAdd(&pred[n * 3 + 2], v2 * inv);
}

// ---------------------------------------------------------------------------
static inline int cdiv(long long a, long long b) { return (int)((a + b - 1) / b); }

extern "C" void kernel_launch(void* const* d_in, const int* in_sizes, int n_in,
                              void* d_out, int out_size, void* d_ws, size_t ws_size,
                              hipStream_t stream) {
    const int*   z    = (const int*)d_in[0];
    const float* t    = (const float*)d_in[3];
    const int*   ei   = (const int*)d_in[4];
    const float* ew   = (const float*)d_in[5];
    const float* ev   = (const float*)d_in[6];
    const float* emb  = (const float*)d_in[7];
    const float* wWt  = (const float*)d_in[8];
    const float* wWq  = (const float*)d_in[9];
    const float* wWk  = (const float*)d_in[10];
    const float* wWv  = (const float*)d_in[11];
    const float* wWdk = (const float*)d_in[12];
    const float* wWdv = (const float*)d_in[13];
    const float* wWo  = (const float*)d_in[14];
    const float* wWvec= (const float*)d_in[15];
    const float* gW[2][5] = {
        {(const float*)d_in[16], (const float*)d_in[17], (const float*)d_in[18],
         (const float*)d_in[19], (const float*)d_in[20]},
        {(const float*)d_in[21], (const float*)d_in[22], (const float*)d_in[23],
         (const float*)d_in[24], (const float*)d_in[25]}};

    float* out = (float*)d_out;
    float* velP = out;
    float* denP = out + (size_t)NN * 3;
    float* hOut = out + (size_t)NN * 6;
    float* vecOut = out + (size_t)NN * 6 + (size_t)NN * 128;

    // ---- workspace bump allocator (256B aligned) ----
    char* base = (char*)d_ws;
    size_t off = 0;
    auto alloc = [&](size_t bytes) -> void* {
        void* p = base + off;
        off = (off + bytes + 255) & ~(size_t)255;
        return p;
    };
    // persistent
    _Float16* rbfH = (_Float16*)alloc((size_t)EE * RR * 2);
    float* ccut = (float*)alloc((size_t)EE * 4);
    float* uvec = (float*)alloc((size_t)EE * 3 * 4);
    float* h    = (float*)alloc((size_t)NN * 128 * 4);
    float* vec  = (float*)alloc((size_t)NN * 384 * 4);
    _Float16* qkvT  = (_Float16*)alloc((size_t)LL * 640 * 128 * 2);
    _Float16* dkvT  = (_Float16*)alloc((size_t)LL * 512 * 64 * 2);
    _Float16* wvecT = (_Float16*)alloc((size_t)LL * 384 * 128 * 2);
    _Float16* woT   = (_Float16*)alloc((size_t)LL * 384 * 128 * 2);
    _Float16* gT[2][4];
    for (int f = 0; f < 2; ++f) {
        gT[f][0] = (_Float16*)alloc((size_t)256 * 128 * 2);  // WghT [256,128]
        gT[f][1] = (_Float16*)alloc((size_t)128 * 384 * 2);  // Ws1T [128,384]
        gT[f][2] = (_Float16*)alloc((size_t)128 * 128 * 2);  // WgT
        gT[f][3] = (_Float16*)alloc((size_t)128 * 256 * 2);  // WmuT [128,256]
    }
    size_t scratchStart = off;
    // encoder-layer temporaries
    _Float16* hnH  = (_Float16*)alloc((size_t)NN * 128 * 2);
    float* qkv  = (float*)alloc((size_t)NN * 640 * 4);
    float* dkvC = (float*)alloc((size_t)CHUNK * 512 * 4);
    float* msgH = (float*)alloc((size_t)NN * 128 * 4);
    float* msgV = (float*)alloc((size_t)NN * 384 * 4);
    _Float16* msgHH = (_Float16*)alloc((size_t)NN * 128 * 2);
    _Float16* vecH  = (_Float16*)alloc((size_t)NN * 384 * 2);
    float* vp = (float*)alloc((size_t)NN * 3 * 384 * 4);
    float* o  = (float*)alloc((size_t)NN * 384 * 4);

    auto gemm = [&](const _Float16* A, const _Float16* Bt, float* C, const float* bias,
                    int M, int K, int Nc, int act) {
        int tiles = (M / 16) * (Nc / 64);
        wmma_gemm_k<<<cdiv(tiles, 8), 256, 0, stream>>>(A, Bt, C, bias, M, K, Nc, act);
    };
    auto tr = [&](const float* W, _Float16* Wt, int K, int Nc) {
        trans_f16_k<<<cdiv((long long)K * Nc, 256), 256, 0, stream>>>(W, Wt, K, Nc);
    };
    auto zero = [&](float* p, long long n) {
        zero_f32_k<<<cdiv(n, 256), 256, 0, stream>>>(p, n);
    };
    auto tohalf = [&](const float* a, _Float16* b, long long n) {
        f32_to_f16_k<<<cdiv(n, 256), 256, 0, stream>>>(a, b, n);
    };

    // ---- weight conversion / transposition (f16) ----
    for (int l = 0; l < LL; ++l) {
        _Float16* qT = qkvT + (size_t)l * 640 * 128;
        tr(wWq + (size_t)l * 128 * 128, qT, 128, 128);
        tr(wWk + (size_t)l * 128 * 128, qT + (size_t)128 * 128, 128, 128);
        tr(wWv + (size_t)l * 128 * 384, qT + (size_t)256 * 128, 128, 384);
        _Float16* dT = dkvT + (size_t)l * 512 * 64;
        tr(wWdk + (size_t)l * 64 * 128, dT, 64, 128);
        tr(wWdv + (size_t)l * 64 * 384, dT + (size_t)128 * 64, 64, 384);
        tr(wWvec + (size_t)l * 128 * 384, wvecT + (size_t)l * 384 * 128, 128, 384);
        tr(wWo   + (size_t)l * 128 * 384, woT   + (size_t)l * 384 * 128, 128, 384);
    }
    for (int f = 0; f < 2; ++f) {
        tr(gW[f][0], gT[f][0], 128, 256);
        tr(gW[f][1], gT[f][1], 384, 128);
        tr(gW[f][3], gT[f][2], 128, 128);
        tr(gW[f][4], gT[f][3], 256, 128);
    }

    // ---- edge features + node init ----
    edge_pre_k<<<cdiv(EE, 256), 256, 0, stream>>>(ew, ev, rbfH, ccut, uvec);
    node_init_k<<<cdiv((long long)NN * 128, 256), 256, 0, stream>>>(z, t, emb, wWt, h);
    zero(vec, (long long)NN * 384);

    // ---- encoder layers ----
    for (int l = 0; l < LL; ++l) {
        ln_k<<<NN, 128, 0, stream>>>(h, hnH);
        gemm(hnH, qkvT + (size_t)l * 640 * 128, qkv, nullptr, NN, 128, 640, 0);
        zero(msgH, (long long)NN * 128);
        zero(msgV, (long long)NN * 384);
        for (int c0 = 0; c0 < EE; c0 += CHUNK) {
            int cnt = (EE - c0 < CHUNK) ? (EE - c0) : CHUNK;
            gemm(rbfH + (size_t)c0 * RR, dkvT + (size_t)l * 512 * 64, dkvC,
                 nullptr, cnt, 64, 512, 1 /*silu*/);
            edge_msg_k<<<cdiv(cnt, 8), 256, 0, stream>>>(ei, ccut, uvec, qkv, dkvC,
                                                         vec, msgH, msgV, c0, cnt);
        }
        tohalf(vec, vecH, (long long)NN * 384);
        gemm(vecH, wvecT + (size_t)l * 384 * 128, vp, nullptr, NN * 3, 128, 384, 0);
        tohalf(msgH, msgHH, (long long)NN * 128);
        gemm(msgHH, woT + (size_t)l * 384 * 128, o, nullptr, NN, 128, 384, 0);
        node_upd_k<<<cdiv((long long)NN * 128, 256), 256, 0, stream>>>(h, vec, vp, o, msgV);
    }

    // ---- GVP FFN heads (re-use scratch region) ----
    off = scratchStart;
    _Float16* vecH2  = (_Float16*)alloc((size_t)NN * 384 * 2);
    float*    vh     = (float*)alloc((size_t)NN * 3 * 256 * 4);
    _Float16* vhH    = (_Float16*)alloc((size_t)NN * 3 * 256 * 2);
    _Float16* catH   = (_Float16*)alloc((size_t)NN * 384 * 2);
    float*    sfeat  = (float*)alloc((size_t)NN * 128 * 4);
    _Float16* sfeatH = (_Float16*)alloc((size_t)NN * 128 * 2);
    float*    gate   = (float*)alloc((size_t)NN * 128 * 4);
    float*    vout   = (float*)alloc((size_t)NN * 384 * 4);

    tohalf(vec, vecH2, (long long)NN * 384);
    for (int f = 0; f < 2; ++f) {
        float* pred = (f == 0) ? velP : denP;
        gemm(vecH2, gT[f][0], vh, nullptr, NN * 3, 128, 256, 0);
        cat_build_k<<<cdiv((long long)NN * 384, 256), 256, 0, stream>>>(h, vh, catH);
        gemm(catH, gT[f][1], sfeat, gW[f][2] /*bs1*/, NN, 384, 128, 1 /*silu*/);
        tohalf(sfeat, sfeatH, (long long)NN * 128);
        gemm(sfeatH, gT[f][2], gate, nullptr, NN, 128, 128, 2 /*sigmoid*/);
        tohalf(vh, vhH, (long long)NN * 3 * 256);
        gemm(vhH, gT[f][3], vout, nullptr, NN * 3, 256, 128, 0);
        zero(pred, (long long)NN * 3);
        gvp_fin_k<<<cdiv((long long)NN * 128, 256), 256, 0, stream>>>(vout, gate, pred);
    }

    // ---- copy h, vec to output ----
    copy_f32_k<<<cdiv((long long)NN * 128, 256), 256, 0, stream>>>(h, hOut, (long long)NN * 128);
    copy_f32_k<<<cdiv((long long)NN * 384, 256), 256, 0, stream>>>(vec, vecOut, (long long)NN * 384);
}
